// Quantumnet_15539191677353
// MI455X (gfx1250) — matrix operand order repaired and checked by reading, not compiled, not run on hardware
//
#include <hip/hip_runtime.h>
#include <hip/hip_bf16.h>

// ---------------------------------------------------------------------------
// QuantumNet on MI455X (gfx1250, wave32, WMMA, async LDS copy).
//
//   k_qin    : 8192x512 @ 512x10 reduction + tanh -> per-wire 2-vectors V
//   k_circuit: build fixed-circuit matrix M (1024x1024) in LDS, store f16
//   k_psi    : materialize rank-1 product states Psi (8192x1024) f16
//   k_gemm   : Phi = Psi @ M^T via v_wmma_f32_16x16x32_f16, double-buffered
//              global_load_async_to_lds_b128 staging, fused Walsh-Hadamard
//              <Z_w> epilogue -> atomic f32 into zsum
//   k_post   : out = zsum @ W_post^T + b_post
// ---------------------------------------------------------------------------

typedef _Float16 v16h __attribute__((ext_vector_type(16)));
typedef _Float16 v8h  __attribute__((ext_vector_type(8)));
typedef float    v8f  __attribute__((ext_vector_type(8)));

// types for the async-copy builtin (signature revealed by clang diagnostic:
// first param is 'int __vector(4) * addrspace(1)', second LDS side)
typedef int v4i_gcc __attribute__((__vector_size__(16)));
typedef __attribute__((address_space(1))) v4i_gcc* as1_v4i;
typedef __attribute__((address_space(3))) v4i_gcc* as3_v4i;
typedef __attribute__((address_space(1))) void* as1_void;
typedef __attribute__((address_space(3))) void* as3_void;

#define NQ   10
#define DIM  1024
#define B_SZ 8192
#define DIN  512

// ---------------------------------------------------------------------------
// Kernel 1: q_in = tanh(X @ W_red^T + b_red) * pi/2, then per-wire 2-vector
//   v0 = (cos(t/2) - sin(t/2)) / sqrt(2), v1 = (cos(t/2) + sin(t/2)) / sqrt(2)
// ---------------------------------------------------------------------------
__global__ __launch_bounds__(128) void k_qin(const float* __restrict__ X,
                                             const float* __restrict__ Wr,
                                             const float* __restrict__ br,
                                             float* __restrict__ V) {
    const int b = blockIdx.x;
    const int t = threadIdx.x;            // 128 threads
    __shared__ float red[128][NQ];

    float acc[NQ];
#pragma unroll
    for (int w = 0; w < NQ; ++w) acc[w] = 0.f;

    const float* xr = X + (size_t)b * DIN;
    for (int e = t; e < DIN; e += 128) {
        const float xv = xr[e];
#pragma unroll
        for (int w = 0; w < NQ; ++w) acc[w] += xv * Wr[w * DIN + e];
    }
#pragma unroll
    for (int w = 0; w < NQ; ++w) red[t][w] = acc[w];
    __syncthreads();
    for (int sft = 64; sft >= 1; sft >>= 1) {
        if (t < sft) {
#pragma unroll
            for (int w = 0; w < NQ; ++w) red[t][w] += red[t + sft][w];
        }
        __syncthreads();
    }
    if (t < NQ) {
        const float z  = red[0][t] + br[t];
        const float th = tanhf(z) * 1.5707963267948966f;
        const float c  = cosf(0.5f * th);
        const float s  = sinf(0.5f * th);
        V[(b * NQ + t) * 2 + 0] = (c - s) * 0.7071067811865476f;
        V[(b * NQ + t) * 2 + 1] = (c + s) * 0.7071067811865476f;
    }
}

// ---------------------------------------------------------------------------
// Kernel 2: fixed-circuit matrix M. Block j pushes basis state e_j through
// 6 layers of (even CNOTs, odd CNOTs, shared-angle RY) in LDS, writes column
// j of M as f16 row-major (M[i][j] = B^T storage for the GEMM).
// ---------------------------------------------------------------------------
__global__ __launch_bounds__(256) void k_circuit(const float* __restrict__ qp,
                                                 _Float16* __restrict__ Mrm) {
    const int j = blockIdx.x;
    const int t = threadIdx.x;            // 256 threads
    __shared__ float st[DIM];

    for (int u = t; u < DIM; u += 256) st[u] = (u == j) ? 1.f : 0.f;
    __syncthreads();

    for (int k = 0; k < 6; ++k) {
        // even CNOTs: ctrl bits {9,7,5,3,1} -> target one below (mask 0x155)
        for (int u = t; u < DIM; u += 256) {
            const int y = u ^ ((u >> 1) & 0x155);
            if (y > u) { const float tmp = st[u]; st[u] = st[y]; st[y] = tmp; }
        }
        __syncthreads();
        // odd CNOTs: ctrl bits {8,6,4,2} -> targets (mask 0x0AA)
        for (int u = t; u < DIM; u += 256) {
            const int y = u ^ ((u >> 1) & 0x0AA);
            if (y > u) { const float tmp = st[u]; st[u] = st[y]; st[y] = tmp; }
        }
        __syncthreads();
        // RY layer, shared angles qw[k+1][w]
        for (int w = 0; w < NQ; ++w) {
            const float th = qp[(k + 1) * NQ + w];
            const float c = cosf(0.5f * th);
            const float s = sinf(0.5f * th);
            const int bp = 9 - w;
            const int bm = 1 << bp;
            for (int p = t; p < DIM / 2; p += 256) {
                const int x0 = ((p >> bp) << (bp + 1)) | (p & (bm - 1));
                const int x1 = x0 | bm;
                const float a0 = st[x0], a1 = st[x1];
                st[x0] = c * a0 - s * a1;
                st[x1] = s * a0 + c * a1;
            }
            __syncthreads();
        }
    }
    for (int u = t; u < DIM; u += 256)
        Mrm[(size_t)u * DIM + j] = (_Float16)st[u];
}

// ---------------------------------------------------------------------------
// Kernel 3: Psi[b][x] = prod_w v[b][w][(x >> (9-w)) & 1], stored f16.
// ---------------------------------------------------------------------------
__global__ __launch_bounds__(256) void k_psi(const float* __restrict__ V,
                                             _Float16* __restrict__ Psi) {
    const int b = blockIdx.x;
    const int t = threadIdx.x;            // 256 threads
    __shared__ float v[NQ][2];
    if (t < 2 * NQ) ((float*)v)[t] = V[b * 2 * NQ + t];
    __syncthreads();
#pragma unroll
    for (int u = 0; u < 4; ++u) {
        const int x = t * 4 + u;
        float p = 1.f;
#pragma unroll
        for (int w = 0; w < NQ; ++w) p *= v[w][(x >> (9 - w)) & 1];
        Psi[(size_t)b * DIM + x] = (_Float16)p;
    }
}

// ---------------------------------------------------------------------------
// Kernel 4: Phi = Psi @ M^T via WMMA + fused <Z> epilogue.
//   block tile 128(b) x 64(i), K-step 64, double-buffered LDS staging via
//   CDNA5 global_load_async_to_lds_b128 (fallback: batched register copy).
// ---------------------------------------------------------------------------
#define BM 128
#define BN 64
#define BK 64
#define LDA 72                             // padded LDS row stride (f16)

__device__ __forceinline__ void stage_tiles(const _Float16* __restrict__ Psi,
                                            const _Float16* __restrict__ Mrm,
                                            _Float16* as, _Float16* bs,
                                            int b0, int i0, int k0, int tid) {
#if __has_builtin(__builtin_amdgcn_global_load_async_to_lds_b128)
    // direct memory -> LDS, no VGPR transit, tracked by ASYNCcnt
#pragma unroll
    for (int cc = 0; cc < 4; ++cc) {       // A: 128x64 f16, 4 chunks/thread
        const int ch = tid + cc * 256;
        const int row = ch >> 3, kc = (ch & 7) * 8;
        __builtin_amdgcn_global_load_async_to_lds_b128(
            (as1_v4i)(as1_void)&Psi[(size_t)(b0 + row) * DIM + k0 + kc],
            (as3_v4i)(as3_void)&as[row * LDA + kc], 0, 0);
    }
#pragma unroll
    for (int cc = 0; cc < 2; ++cc) {       // B: 64x64 f16, 2 chunks/thread
        const int ch = tid + cc * 256;
        const int row = ch >> 3, kc = (ch & 7) * 8;
        __builtin_amdgcn_global_load_async_to_lds_b128(
            (as1_v4i)(as1_void)&Mrm[(size_t)(i0 + row) * DIM + k0 + kc],
            (as3_v4i)(as3_void)&bs[row * LDA + kc], 0, 0);
    }
#else
    uint4 ta[4], tb[2];
#pragma unroll
    for (int cc = 0; cc < 4; ++cc) {
        const int ch = tid + cc * 256;
        const int row = ch >> 3, kc = (ch & 7) * 8;
        ta[cc] = *(const uint4*)&Psi[(size_t)(b0 + row) * DIM + k0 + kc];
    }
#pragma unroll
    for (int cc = 0; cc < 2; ++cc) {
        const int ch = tid + cc * 256;
        const int row = ch >> 3, kc = (ch & 7) * 8;
        tb[cc] = *(const uint4*)&Mrm[(size_t)(i0 + row) * DIM + k0 + kc];
    }
#pragma unroll
    for (int cc = 0; cc < 4; ++cc) {
        const int ch = tid + cc * 256;
        const int row = ch >> 3, kc = (ch & 7) * 8;
        *(uint4*)&as[row * LDA + kc] = ta[cc];
    }
#pragma unroll
    for (int cc = 0; cc < 2; ++cc) {
        const int ch = tid + cc * 256;
        const int row = ch >> 3, kc = (ch & 7) * 8;
        *(uint4*)&bs[row * LDA + kc] = tb[cc];
    }
#endif
}

__device__ __forceinline__ void wait_async() {
#if __has_builtin(__builtin_amdgcn_s_wait_asynccnt)
    __builtin_amdgcn_s_wait_asynccnt(0);
#elif __has_builtin(__builtin_amdgcn_global_load_async_to_lds_b128)
    asm volatile("s_wait_asynccnt 0x0" ::: "memory");
#endif
}

__global__ __launch_bounds__(256) void k_gemm(const _Float16* __restrict__ Psi,
                                              const _Float16* __restrict__ Mrm,
                                              float* __restrict__ zsum) {
    __shared__ _Float16 As[2][BM * LDA];
    __shared__ _Float16 Bs[2][BN * LDA];

    const int i0   = blockIdx.x * BN;
    const int b0   = blockIdx.y * BM;
    const int tid  = threadIdx.x;
    const int lane = tid & 31;
    const int wv   = tid >> 5;
    const int n    = lane & 15;
    const int q    = lane >> 4;

    v8f acc[4];
#pragma unroll
    for (int tt = 0; tt < 4; ++tt) {
        v8f z = {0.f, 0.f, 0.f, 0.f, 0.f, 0.f, 0.f, 0.f};
        acc[tt] = z;
    }

    stage_tiles(Psi, Mrm, As[0], Bs[0], b0, i0, 0, tid);

    const int NIT = DIM / BK;              // 16
    for (int it = 0; it < NIT; ++it) {
        wait_async();                      // this wave's fills done
        __syncthreads();                   // everyone's fills done + prior
                                           // compute on the other buffer done
        const _Float16* asb = As[it & 1];
        const _Float16* bsb = Bs[it & 1];
        if (it + 1 < NIT)                  // prefetch next K-slice
            stage_tiles(Psi, Mrm, As[(it + 1) & 1], Bs[(it + 1) & 1],
                        b0, i0, (it + 1) * BK, tid);

#pragma unroll
        for (int kk = 0; kk < BK; kk += 32) {
            // A-frag 16x32: half q selects K {kk..kk+15} / {kk+16..kk+31}
            const int arow = wv * 16 + n;
            const v8h alo = *(const v8h*)&asb[arow * LDA + kk + 16 * q];
            const v8h ahi = *(const v8h*)&asb[arow * LDA + kk + 16 * q + 8];
            const v16h af = __builtin_shufflevector(
                alo, ahi, 0, 1, 2, 3, 4, 5, 6, 7, 8, 9, 10, 11, 12, 13, 14, 15);
            // B-frags 32x16: half q selects K {kk..kk+15} / {kk+16..kk+31}
            v16h bf[4];
#pragma unroll
            for (int tt = 0; tt < 4; ++tt) {
                const int brow = tt * 16 + n;
                const v8h blo = *(const v8h*)&bsb[brow * LDA + kk + 16 * q];
                const v8h bhi = *(const v8h*)&bsb[brow * LDA + kk + 16 * q + 8];
                bf[tt] = __builtin_shufflevector(
                    blo, bhi, 0, 1, 2, 3, 4, 5, 6, 7, 8, 9, 10, 11, 12, 13, 14, 15);
            }
#pragma unroll
            for (int tt = 0; tt < 4; ++tt)
                acc[tt] = __builtin_amdgcn_wmma_f32_16x16x32_f16(
                    false, af, false, bf[tt], (short)0, acc[tt], false, false);
        }
    }

    // ---- fused <Z> epilogue -----------------------------------------------
    // col i = i0 + tt*16 + n. Wire w reads bit (9-w) of i:
    //   w=0..3 -> i0 bits 9..6 (block-uniform sign on total T)
    //   w=4,5  -> tt bits 1,0  (D5, D4 tile-sum combos, plain lane reduce)
    //   w=6..9 -> n  bits 3..0 (single-bit rows of 16-lane Walsh-Hadamard)
    float sg[4];
#pragma unroll
    for (int w = 0; w < 4; ++w) sg[w] = ((i0 >> (9 - w)) & 1) ? -1.f : 1.f;

#pragma unroll
    for (int r = 0; r < 8; ++r) {
        const float p0 = acc[0][r] * acc[0][r];
        const float p1 = acc[1][r] * acc[1][r];
        const float p2 = acc[2][r] * acc[2][r];
        const float p3 = acc[3][r] * acc[3][r];
        float T  = (p0 + p1) + (p2 + p3);
        float D5 = (p0 + p1) - (p2 + p3);  // (-1)^{bit5 of i} = tt>=2
        float D4 = (p0 - p1) + (p2 - p3);  // (-1)^{bit4 of i} = tt&1
        float h  = T;
#pragma unroll
        for (int bmask = 1; bmask <= 8; bmask <<= 1) {
            const float o = __shfl_xor(h, bmask, 32);
            h = (n & bmask) ? (o - h) : (o + h);
            D5 += __shfl_xor(D5, bmask, 32);
            D4 += __shfl_xor(D4, bmask, 32);
        }
        // h on lane n==mask holds sum_n (-1)^{popcnt(n&mask)} T
        const int row = b0 + wv * 16 + r + 8 * q;
        float* zr = &zsum[row * NQ];
        if (n == 0) {
            atomicAdd(zr + 0, sg[0] * h);  // h(0) = total
            atomicAdd(zr + 1, sg[1] * h);
            atomicAdd(zr + 2, sg[2] * h);
            atomicAdd(zr + 3, sg[3] * h);
            atomicAdd(zr + 4, D5);
            atomicAdd(zr + 5, D4);
        } else if (n == 8) { atomicAdd(zr + 6, h);
        } else if (n == 4) { atomicAdd(zr + 7, h);
        } else if (n == 2) { atomicAdd(zr + 8, h);
        } else if (n == 1) { atomicAdd(zr + 9, h);
        }
    }
}

// ---------------------------------------------------------------------------
// Kernel 5: out = zsum @ W_post^T + b_post   (8192 x 2)
// ---------------------------------------------------------------------------
__global__ __launch_bounds__(256) void k_post(const float* __restrict__ zsum,
                                              const float* __restrict__ Wp,
                                              const float* __restrict__ bp,
                                              float* __restrict__ out) {
    const int b = blockIdx.x * blockDim.x + threadIdx.x;
    if (b >= B_SZ) return;
    float z[NQ];
#pragma unroll
    for (int w = 0; w < NQ; ++w) z[w] = zsum[b * NQ + w];
#pragma unroll
    for (int c = 0; c < 2; ++c) {
        float a = bp[c];
#pragma unroll
        for (int w = 0; w < NQ; ++w) a += z[w] * Wp[c * NQ + w];
        out[b * 2 + c] = a;
    }
}

// ---------------------------------------------------------------------------
extern "C" void kernel_launch(void* const* d_in, const int* in_sizes, int n_in,
                              void* d_out, int out_size, void* d_ws, size_t ws_size,
                              hipStream_t stream) {
    (void)in_sizes; (void)n_in; (void)out_size; (void)ws_size;
    const float* X  = (const float*)d_in[0];   // input_features 8192x512
    const float* Wr = (const float*)d_in[1];   // W_red 10x512
    const float* br = (const float*)d_in[2];   // b_red 10
    const float* qp = (const float*)d_in[3];   // q_params 150
    const float* Wp = (const float*)d_in[4];   // W_post 2x10
    const float* bp = (const float*)d_in[5];   // b_post 2
    float* out = (float*)d_out;

    char* ws = (char*)d_ws;
    const size_t V_OFF    = 0;                               // 8192*10*2 f32
    const size_t M_OFF    = V_OFF + (size_t)B_SZ * NQ * 2 * 4;
    const size_t PSI_OFF  = M_OFF + (size_t)DIM * DIM * 2;   // f16
    const size_t ZSUM_OFF = PSI_OFF + (size_t)B_SZ * DIM * 2;
    float*     V    = (float*)(ws + V_OFF);
    _Float16*  Mrm  = (_Float16*)(ws + M_OFF);
    _Float16*  Psi  = (_Float16*)(ws + PSI_OFF);
    float*     zsum = (float*)(ws + ZSUM_OFF);

    (void)hipMemsetAsync(zsum, 0, (size_t)B_SZ * NQ * 4, stream);

    k_qin    <<<B_SZ, 128, 0, stream>>>(X, Wr, br, V);
    k_circuit<<<DIM, 256, 0, stream>>>(qp, Mrm);
    k_psi    <<<B_SZ, 256, 0, stream>>>(V, Psi);

    dim3 grid(DIM / BN, B_SZ / BM);        // (16, 64)
    k_gemm   <<<grid, 256, 0, stream>>>(Psi, Mrm, zsum);

    k_post   <<<(B_SZ + 255) / 256, 256, 0, stream>>>(zsum, Wp, bp, out);
}
